// LFQ_21182778704823
// MI455X (gfx1250) — matrix operand order repaired and checked by reading, not compiled, not run on hardware
//
#include <hip/hip_runtime.h>
#include <math.h>

typedef __attribute__((ext_vector_type(16))) _Float16 v16h;
typedef __attribute__((ext_vector_type(8)))  float    v8f;

#define BATCH 256
#define CV    16
#define NPIX  1024            // 32*32
#define NELEM (CV*NPIX)       // 16384
#define SN    6
#define VOCABN 65536
#define EPSF  1e-5f

__device__ __forceinline__ float cubic_w(float t) {
    const float a = -0.75f;
    t = fabsf(t);
    float t2 = t * t, t3 = t2 * t;
    if (t <= 1.f) return (a + 2.f) * t3 - (a + 3.f) * t2 + 1.f;
    if (t < 2.f)  return a * t3 - 5.f * a * t2 + 8.f * a * t - 4.f * a;
    return 0.f;
}

// One workgroup (256 threads = 8 waves) per batch image. Whole residual field
// lives in LDS; HBM touched once in, once out.
extern "C" __global__ __launch_bounds__(256)
void lfq_main(const float* __restrict__ f,
              const float* __restrict__ phi_w,   // (4, O=16, I=16, 3, 3)
              const float* __restrict__ phi_b,   // (4, 16)
              float* __restrict__ fhat_out,      // (256,16,32,32)
              float* __restrict__ vocab_hit,     // (6, 65536)
              float* __restrict__ ws)            // msq[6], ent[6], psum[6*16]
{
    extern __shared__ char smem[];
    float*    s_rest  = (float*)smem;             // 16384 f  (residual, == x)
    float*    s_h     = s_rest + NELEM;           // 16384 f  (h field, pre-phi)
    float*    s_small = s_h + NELEM;              // 4096  f  (area-down result)
    _Float16* s_bmat  = (_Float16*)(s_small + 4096); // 160x16 f16 weight GEMM B
    float*    s_red   = (float*)(s_bmat + 2560);  // 32 f reduction slots

    const int tid  = threadIdx.x;
    const int b    = blockIdx.x;
    const int lane = tid & 31;
    const int wave = tid >> 5;

    const float* fb = f + (size_t)b * NELEM;
    __builtin_prefetch(fb + tid * 16, 0, 1);      // global_prefetch_b8

    // load f -> s_rest (f_hat starts at 0, so rest == f)
    {
        const float4* f4 = (const float4*)fb;
        float4* r4 = (float4*)s_rest;
        for (int i = tid; i < NELEM / 4; i += 256) r4[i] = f4[i];
    }
    __syncthreads();

    float* ws_msq = ws;
    float* ws_ent = ws + SN;
    float* ws_ps  = ws + 2 * SN;

    for (int si = 0; si < SN; ++si) {
        const int pn   = 1 << si;
        const int phi  = (si <= 1) ? 0 : ((si >= 4) ? 3 : si - 1); // PHI_IDX
        const int blk  = 32 / pn;
        const float invb = 1.0f / (float)(blk * blk);
        const int npx  = pn * pn;

        if (tid < 32) s_red[tid] = 0.0f;

        // ---- A: area-downsample rest -> s_small (skip at last scale) ----
        if (si < SN - 1) {
            const int ncell = CV * npx;
            for (int cell = tid; cell < ncell; cell += 256) {
                int c  = cell / npx;
                int r  = cell - c * npx;
                int ys = r / pn, xs = r - ys * pn;
                const float* base = s_rest + c * NPIX + (ys * blk) * 32 + xs * blk;
                float s = 0.f;
                for (int yy = 0; yy < blk; ++yy)
                    for (int xx = 0; xx < blk; ++xx)
                        s += base[yy * 32 + xx];
                s_small[cell] = s * invb;
            }
        }
        __syncthreads();

        // ---- B: sign-pack 16 channels -> idx, histogram ----
        for (int j = tid; j < npx; j += 256) {
            unsigned idx = 0u;
            for (int c = 0; c < CV; ++c) {
                float v = (si < SN - 1) ? s_small[c * npx + j]
                                        : s_rest[c * NPIX + j];
                idx |= (v > 0.f ? 1u : 0u) << c;
            }
            atomicAdd(&vocab_hit[(size_t)si * VOCABN + idx], 1.0f);
        }

        // ---- C: entropy terms on x = rest (before update) ----
        {
            float es_tot = 0.f;
            for (int c = 0; c < CV; ++c) {
                float ps = 0.f;
                for (int r = tid; r < NPIX; r += 256) {
                    float x = s_rest[c * NPIX + r];
                    float p = 1.0f / (1.0f + __expf(400.0f * x)); // sigmoid(-400x)
                    float q = 1.0f - p;
                    ps += p;
                    es_tot -= p * __logf(p + EPSF) + q * __logf(q + EPSF);
                }
                atomicAdd(&s_red[c], ps);       // ds_add_f32
            }
            atomicAdd(&s_red[16], es_tot);
        }

        // ---- D: h = bicubic_up(sign(rest_small))  (last scale: direct sign) ----
        if (si < SN - 1) {
            for (int e = tid; e < NELEM; e += 256) {
                int c = e >> 10, pidx = e & 1023;
                int y = pidx >> 5, x = pidx & 31;
                float fy = (y + 0.5f) * pn * (1.0f / 32.0f) - 0.5f;
                float fx = (x + 0.5f) * pn * (1.0f / 32.0f) - 0.5f;
                int y0 = (int)floorf(fy), x0 = (int)floorf(fx);
                float acc = 0.f;
                for (int ky = 0; ky < 4; ++ky) {
                    int jy = y0 - 1 + ky;
                    float wy = cubic_w(fy - (float)jy);
                    int jyc = min(max(jy, 0), pn - 1);
                    for (int kx = 0; kx < 4; ++kx) {
                        int jx = x0 - 1 + kx;
                        float wx = cubic_w(fx - (float)jx);
                        int jxc = min(max(jx, 0), pn - 1);
                        float v = s_small[c * npx + jyc * pn + jxc];
                        acc += wy * wx * (v > 0.f ? 1.f : -1.f);
                    }
                }
                s_h[e] = acc;
            }
        } else {
            for (int e = tid; e < NELEM; e += 256)
                s_h[e] = (s_rest[e] > 0.f) ? 1.f : -1.f;
        }

        // ---- E: stage GEMM B matrix (K=160 x N=16, zero-padded) as fp16 ----
        for (int t = tid; t < 160 * 16; t += 256) {
            int k = t >> 4, n = t & 15;
            float w = 0.f;
            if (k < 144) {
                int ci = k / 9;
                int r  = k - ci * 9;            // kh*3+kw
                w = phi_w[(((phi * CV + n) * CV + ci) * 9) + r];
            }
            s_bmat[t] = (_Float16)w;
        }
        __syncthreads();

        // flush entropy partials (one global atomic per channel per block)
        if (tid < 16) atomicAdd(&ws_ps[si * 16 + tid], s_red[tid]);
        if (tid == 16) atomicAdd(&ws_ent[si], s_red[16]);

        // ---- F: conv3x3 as WMMA GEMM (im2col K=144), phi blend, rest update ----
        {
            // B fragments: lane layout  halves i -> K = ks*32 + 16*(lane>>4) + i,
            //                           N = lane&15   (dense f16 B 32x16)
            v16h bf[5];
            #pragma unroll
            for (int ks = 0; ks < 5; ++ks) {
                int kbase = ks * 32 + ((lane >> 4) << 4);
                const _Float16* bp = s_bmat + kbase * 16 + (lane & 15);
                v16h tb;
                #pragma unroll
                for (int i = 0; i < 16; ++i) tb[i] = bp[i * 16];
                bf[ks] = tb;
            }
            const float bco = phi_b[phi * CV + (lane & 15)];
            float msq = 0.f;
            for (int t = wave; t < 64; t += 8) {          // 64 pixel-tiles of 16
                const int pbase = t * 16;
                const int prow  = pbase + (lane & 15);    // A row = pixel
                const int py = prow >> 5, px = prow & 31;
                v8f acc = {};
                #pragma unroll
                for (int ks = 0; ks < 5; ++ks) {
                    // A halves i -> K = ks*32 + 16*(lane>>4) + i
                    v16h af;
                    #pragma unroll
                    for (int i = 0; i < 16; ++i) {
                        int K = ks * 32 + ((lane >> 4) << 4) + i;
                        float v = 0.f;
                        if (K < 144) {
                            int ci = K / 9;
                            int r  = K - ci * 9;
                            int dy = r / 3 - 1, dx = (r - (r / 3) * 3) - 1;
                            int yy = py + dy, xx = px + dx;
                            if ((unsigned)yy < 32u && (unsigned)xx < 32u)
                                v = s_h[ci * NPIX + yy * 32 + xx];  // SAME, zero pad
                        }
                        af[i] = (_Float16)v;
                    }
                    acc = __builtin_amdgcn_wmma_f32_16x16x32_f16(
                        false, af, false, bf[ks], (short)0, acc, false, false);
                }
                // D layout: VGPR j, lane -> M = j + 8*(lane>>4), N = lane&15
                const int co   = lane & 15;
                const int poff = pbase + ((lane >> 4) << 3);
                #pragma unroll
                for (int j = 0; j < 8; ++j) {
                    int p = poff + j;
                    float hv  = s_h[co * NPIX + p];
                    float out = 0.5f * hv + 0.5f * (acc[j] + bco); // Phi, r=0.5
                    int ei = co * NPIX + p;
                    float nr = s_rest[ei] - out;   // rest -= phi(h)
                    s_rest[ei] = nr;
                    msq += nr * nr;                // (f_hat - f)^2 == rest^2
                }
            }
            atomicAdd(&s_red[24], msq);
        }
        __syncthreads();
        if (tid == 0) atomicAdd(&ws_msq[si], s_red[24]);
        __syncthreads();
    }

    // f_hat_out = f - rest_final (straight-through == f_hat numerically)
    {
        const float4* f4 = (const float4*)fb;
        const float4* r4 = (const float4*)s_rest;
        float4* o4 = (float4*)(fhat_out + (size_t)b * NELEM);
        for (int i = tid; i < NELEM / 4; i += 256) {
            float4 a = f4[i], r = r4[i];
            float4 o;
            o.x = a.x - r.x; o.y = a.y - r.y; o.z = a.z - r.z; o.w = a.w - r.w;
            o4[i] = o;
        }
    }
}

extern "C" __global__ void lfq_finalize(const float* __restrict__ ws,
                                        float* __restrict__ total_out)
{
    if (threadIdx.x != 0 || blockIdx.x != 0) return;
    const float* msq = ws;
    const float* ent = ws + SN;
    const float* ps  = ws + 2 * SN;
    const float nel  = (float)BATCH * (float)CV * (float)NPIX;   // B*C*H*W
    const float nbhw = (float)BATCH * (float)NPIX;               // B*H*W
    float total = 0.f;
    for (int si = 0; si < SN; ++si) {
        total += 1.25f * (msq[si] / nel) * (1.0f / 6.0f);        // vq + commit
        float samp = ent[si] / nbhw;                              // per-sample H
        float cb = 0.f;
        for (int c = 0; c < CV; ++c) {                            // codebook H
            float q  = ps[si * 16 + c] / nbhw;
            float q1 = 1.f - q;
            cb -= q * __logf(q + EPSF) + q1 * __logf(q1 + EPSF);
        }
        total += (0.1f / 6.0f) * (samp - cb);
    }
    *total_out = total;
}

extern "C" void kernel_launch(void* const* d_in, const int* in_sizes, int n_in,
                              void* d_out, int out_size, void* d_ws, size_t ws_size,
                              hipStream_t stream) {
    (void)in_sizes; (void)n_in; (void)out_size; (void)ws_size;
    const float* f     = (const float*)d_in[0];
    const float* phi_w = (const float*)d_in[1];
    const float* phi_b = (const float*)d_in[2];

    float* out   = (float*)d_out;
    float* fhat  = out;                                 // 256*16*32*32
    float* total = out + (size_t)BATCH * NELEM;         // 1
    float* vocab = total + 1;                           // 6*65536
    float* ws    = (float*)d_ws;                        // 108 accumulators

    // Accumulators must be zero every call (harness does not re-poison).
    hipMemsetAsync(ws, 0, 128 * sizeof(float), stream);
    hipMemsetAsync(total, 0, (size_t)(1 + SN * VOCABN) * sizeof(float), stream);

    const size_t smem = (size_t)(NELEM + NELEM + 4096 + 32) * sizeof(float)
                      + (size_t)2560 * sizeof(_Float16);   // ~149 KB of 320 KB WGP LDS
    lfq_main<<<BATCH, 256, smem, stream>>>(f, phi_w, phi_b, fhat, vocab, ws);
    lfq_finalize<<<1, 1, 0, stream>>>(ws, total);
}
